// model_3453153706017
// MI455X (gfx1250) — compile-verified, matrix operand order checked by reading, not run on hardware
//
#include <hip/hip_runtime.h>
#include <hip/hip_bf16.h>
#include <math.h>

typedef __attribute__((ext_vector_type(16))) _Float16 v16h;
typedef __attribute__((ext_vector_type(8)))  float    v8f;
typedef __attribute__((ext_vector_type(2)))  float    v2f;

#define B_    16
#define F_    64
#define H1_   14
#define W1_   254
#define P_    3556      /* H1_*W1_ */
#define HP_   19        /* padded height  (2+14+3) */
#define WP_   259       /* padded width   (2+254+3) */
#define CPP_  4921      /* HP_*WP_ */
#define TS_   56896     /* B_*P_   */
#define K2_   2304      /* F_*36   */
#define KB2_  72        /* K2_/32  */
#define K3_   128
#define KB3_  4
#define N_    28448     /* F_*P_/8 */
#define E_    8
#define OD_   16

// ---------------- weight pre-swizzle into WMMA A-fragment order ----------------
// dest[ ((mt*kbs + kb)*32 + lane)*16 + (g*2+j) ] = w[m*lda + k]
// m = mt*16 + lane%16 ; k = kb*32 + (g>=4?16:0) + (lane>=16?8:0) + (g&3)*2 + j
__global__ void k_prep(const float* __restrict__ wsrc, _Float16* __restrict__ adst,
                       int kbs, int lda) {
  int idx  = blockIdx.x * 256 + threadIdx.x;
  int j2   = idx & 15;
  int lane = (idx >> 4) & 31;
  int kb   = (idx >> 9) % kbs;
  int mt   = (idx >> 9) / kbs;
  int g = j2 >> 1, j = j2 & 1;
  int m = mt * 16 + (lane & 15);
  int k = kb * 32 + ((g >= 4) ? 16 : 0) + ((lane >= 16) ? 8 : 0) + (g & 3) * 2 + j;
  adst[idx] = (_Float16)wsrc[m * lda + k];
}

__global__ void k_zero(float* __restrict__ p, int n) {
  int t = blockIdx.x * 256 + threadIdx.x;
  if (t < n) p[t] = 0.f;
}

// ------- conv1: direct 6x6 stride-2, relu -> f16 into PADDED (19x259) layout ---
__global__ void k_conv1(const float* __restrict__ x, const float* __restrict__ w1,
                        const float* __restrict__ b1, _Float16* __restrict__ h16p) {
  int tid = blockIdx.x * 256 + threadIdx.x;      // exact: B_*F_*P_ = 14224*256
  int ox = tid % W1_;
  int r1 = tid / W1_;
  int oy = r1 % H1_;
  int r2 = r1 / H1_;
  int f  = r2 % F_;
  int b  = r2 / F_;
  float acc = b1[f];
  #pragma unroll
  for (int dy = 0; dy < 6; ++dy)
    #pragma unroll
    for (int dx = 0; dx < 6; ++dx)
      acc += x[(b * 32 + 2 * oy + dy) * 512 + 2 * ox + dx] * w1[f * 36 + dy * 6 + dx];
  acc = acc > 0.f ? acc : 0.f;
  h16p[((b * F_ + f) * HP_ + oy + 2) * WP_ + ox + 2] = (_Float16)acc;
}

// ---- conv2: implicit GEMM, M=64 K=2304 N=TS_, WMMA f16, padded-input gather ---
__global__ void k_conv2(const _Float16* __restrict__ a2, const _Float16* __restrict__ h16p,
                        const float* __restrict__ b2, _Float16* __restrict__ yr16) {
  int lane = threadIdx.x & 31;
  int wave = threadIdx.x >> 5;
  int tile = blockIdx.x * 8 + wave;              // 14224 tiles exact
  int mt = tile & 3;
  int pt = tile >> 2;
  int l16 = lane & 15;
  int hi8 = (lane >= 16) ? 8 : 0;
  int p = pt * 16 + l16;
  int bimg = p / P_;
  int rem  = p - bimg * P_;
  int oy = rem / W1_;
  int ox = rem - oy * W1_;
  int cb = bimg * (F_ * CPP_) + oy * WP_ + ox;   // padded-coords base for this column
  v8f acc = {};
  #pragma unroll 2
  for (int kb = 0; kb < KB2_; ++kb) {
    v16h a = ((const v16h*)a2)[(mt * KB2_ + kb) * 32 + lane];
    v16h bm;
    int kstart = kb * 32 + hi8;
    int ci0 = kstart / 36;
    int t0  = kstart - ci0 * 36;
    #pragma unroll
    for (int g = 0; g < 8; ++g) {
      #pragma unroll
      for (int j = 0; j < 2; ++j) {
        int off = ((g >= 4) ? 16 : 0) + (g & 3) * 2 + j;
        int t = t0 + off;
        int c = ci0;
        if (t >= 36) { c += 1; t -= 36; }        // single wrap: t0<=35, off<=23
        int dy = t / 6;
        int dx = t - dy * 6;
        bm[g * 2 + j] = h16p[cb + c * CPP_ + dy * WP_ + dx];   // no bounds check
      }
    }
    acc = __builtin_amdgcn_wmma_f32_16x16x32_f16(false, a, false, bm, (short)0, acc, false, false);
  }
  #pragma unroll
  for (int g = 0; g < 8; ++g) {
    int m = mt * 16 + g + hi8;
    float v = acc[g] + b2[m];
    v = v > 0.f ? v : 0.f;                       // relu (concat branch)
    yr16[(bimg * F_ + m) * P_ + rem] = (_Float16)v;
  }
}

// ---------------- conv3: 1x1 conv as GEMM, M=64 K=128 N=TS_, WMMA f16 ---------
__global__ void k_conv3(const _Float16* __restrict__ a3, const _Float16* __restrict__ h16p,
                        const _Float16* __restrict__ yr16, const float* __restrict__ b3,
                        float* __restrict__ h3) {
  int lane = threadIdx.x & 31;
  int wave = threadIdx.x >> 5;
  int tile = blockIdx.x * 8 + wave;
  int mt = tile & 3;
  int pt = tile >> 2;
  int l16 = lane & 15;
  int hi8 = (lane >= 16) ? 8 : 0;
  int p = pt * 16 + l16;
  int bimg = p / P_;
  int rem  = p - bimg * P_;
  int oy   = rem / W1_;
  int hoff = bimg * (F_ * CPP_) + rem + oy * 5 + (2 * WP_ + 2);  // padded addr of (oy,ox)
  v8f acc = {};
  for (int kb = 0; kb < KB3_; ++kb) {
    v16h a = ((const v16h*)a3)[(mt * KB3_ + kb) * 32 + lane];
    v16h bm;
    int kbase = kb * 32 + hi8;
    #pragma unroll
    for (int g = 0; g < 8; ++g) {
      #pragma unroll
      for (int j = 0; j < 2; ++j) {
        int k = kbase + ((g >= 4) ? 16 : 0) + (g & 3) * 2 + j;
        _Float16 val = (k < 64) ? h16p[hoff + k * CPP_]
                                : yr16[(bimg * F_ + (k - 64)) * P_ + rem];
        bm[g * 2 + j] = val;
      }
    }
    acc = __builtin_amdgcn_wmma_f32_16x16x32_f16(false, a, false, bm, (short)0, acc, false, false);
  }
  #pragma unroll
  for (int g = 0; g < 8; ++g) {
    int m = mt * 16 + g + hi8;
    h3[(bimg * F_ + m) * P_ + rem] = acc[g] + b3[m];   // no relu after conv3
  }
}

// ---------------- primary caps: u = u_in @ pc_w + pc_b (tiny, VALU) -----------
__global__ void k_pc(const float* __restrict__ h3, const float* __restrict__ pcw,
                     const float* __restrict__ pcb, float* __restrict__ u) {
  int tid = blockIdx.x * 256 + threadIdx.x;      // exact: B_*N_ = 1778*256
  int b = tid / N_;
  int n = tid - b * N_;
  float ui[8];
  #pragma unroll
  for (int i = 0; i < 8; ++i) ui[i] = h3[b * (F_ * P_) + n * 8 + i];
  #pragma unroll
  for (int o = 0; o < 8; ++o) {
    float acc = pcb[n * 8 + o];
    #pragma unroll
    for (int i = 0; i < 8; ++i) acc += ui[i] * pcw[(n * 8 + i) * 8 + o];
    u[(b * N_ + n) * 8 + o] = acc;
  }
}

// ------- u_hat: per-n GEMM (128x8)@(8x16) with f32 WMMA 16x16x4, store f16 ----
__global__ void k_uhat(const float* __restrict__ ecw, const float* __restrict__ u,
                       _Float16* __restrict__ uh) {
  int lane = threadIdx.x & 31;
  int wave = threadIdx.x >> 5;
  int n = blockIdx.x * 8 + wave;                 // exact: 3556*8 = N_
  int l16 = lane & 15;
  int hi  = (lane >= 16) ? 2 : 0;
  // B fragments (u_n^T : 8 x 16 over batch), independent of e
  const float* up = u + ((long)l16 * N_ + n) * 8;
  v2f bf0, bf1;
  bf0[0] = up[hi];     bf0[1] = up[hi + 1];
  bf1[0] = up[4 + hi]; bf1[1] = up[4 + hi + 1];
  for (int e = 0; e < 8; ++e) {
    const float* ap = ecw + (((long)n * 8 + e) * 16 + l16) * 8;
    v2f a0, a1;
    a0[0] = ap[hi];     a0[1] = ap[hi + 1];
    a1[0] = ap[4 + hi]; a1[1] = ap[4 + hi + 1];
    v8f d = {};
    d = __builtin_amdgcn_wmma_f32_16x16x4_f32(false, a0, false, bf0, (short)0, d, false, false);
    d = __builtin_amdgcn_wmma_f32_16x16x4_f32(false, a1, false, bf1, (short)0, d, false, false);
    #pragma unroll
    for (int g = 0; g < 8; ++g) {
      int o = g + ((lane >= 16) ? 8 : 0);
      uh[(((long)l16 * N_ + n) * 8 + e) * 16 + o] = (_Float16)d[g];
    }
  }
}

// ---------------- routing: partial s = sum_n softmax(b)*u_hat -----------------
__global__ void k_spart(const float* __restrict__ bl, const _Float16* __restrict__ uh,
                        float* __restrict__ s) {
  __shared__ float red[256 * 16];
  int b     = blockIdx.x / 28;
  int chunk = blockIdx.x % 28;
  int t = threadIdx.x;
  int e = t & 7;
  int slot = t >> 3;
  float acc[16];
  #pragma unroll
  for (int o = 0; o < 16; ++o) acc[o] = 0.f;
  for (int it = 0; it < 32; ++it) {
    int n = chunk * 1024 + it * 32 + slot;
    if (n < N_) {
      const float* blp = bl + ((long)b * N_ + n) * 8;
      float mx = blp[0];
      #pragma unroll
      for (int j = 1; j < 8; ++j) mx = fmaxf(mx, blp[j]);
      float den = 0.f;
      #pragma unroll
      for (int j = 0; j < 8; ++j) den += __expf(blp[j] - mx);
      float c = __expf(blp[e] - mx) / den;
      v16h uv = *(const v16h*)(uh + (((long)b * N_ + n) * 8 + e) * 16);
      #pragma unroll
      for (int o = 0; o < 16; ++o) acc[o] += c * (float)uv[o];
    }
  }
  #pragma unroll
  for (int o = 0; o < 16; ++o) red[t * 16 + o] = acc[o];
  __syncthreads();
  for (int off = 128; off >= 8; off >>= 1) {   // offsets all multiples of 8 -> same e
    if (t < off) {
      #pragma unroll
      for (int o = 0; o < 16; ++o) red[t * 16 + o] += red[(t + off) * 16 + o];
    }
    __syncthreads();
  }
  if (t < 8) {
    #pragma unroll
    for (int o = 0; o < 16; ++o) atomicAdd(&s[(b * 8 + t) * 16 + o], red[t * 16 + o]);
  }
}

// ---------------- squash (+ final norms) --------------------------------------
__global__ void k_squash(const float* __restrict__ s, float* __restrict__ v,
                         float* __restrict__ out, int last) {
  int t = threadIdx.x;                           // 128 = B_*E_
  const float* sp = s + t * 16;
  float n2 = 0.f;
  #pragma unroll
  for (int o = 0; o < 16; ++o) n2 += sp[o] * sp[o];
  float nrm = sqrtf(n2);
  float scale = n2 / (1.f + n2) / (nrm + 1e-8f);
  float vn2 = 0.f;
  #pragma unroll
  for (int o = 0; o < 16; ++o) {
    float vv = scale * sp[o];
    v[t * 16 + o] = vv;
    vn2 += vv * vv;
  }
  if (last) out[t] = sqrtf(vn2);
}

// ---------------- b_log += <u_hat, v> -----------------------------------------
__global__ void k_bupd(const _Float16* __restrict__ uh, const float* __restrict__ v,
                       float* __restrict__ bl) {
  int tid = blockIdx.x * 256 + threadIdx.x;      // exact: B_*N_*E_ = 14224*256
  int e  = tid & 7;
  int bn = tid >> 3;
  int b  = bn / N_;
  v16h uv = *(const v16h*)(uh + (long)tid * 16);
  const float* vp = v + (b * 8 + e) * 16;
  float d = 0.f;
  #pragma unroll
  for (int o = 0; o < 16; ++o) d += (float)uv[o] * vp[o];
  bl[tid] += d;
}

extern "C" void kernel_launch(void* const* d_in, const int* in_sizes, int n_in,
                              void* d_out, int out_size, void* d_ws, size_t ws_size,
                              hipStream_t stream) {
  const float* x   = (const float*)d_in[0];
  const float* w1  = (const float*)d_in[1];
  const float* b1  = (const float*)d_in[2];
  const float* w2  = (const float*)d_in[3];
  const float* b2  = (const float*)d_in[4];
  const float* w3  = (const float*)d_in[5];
  const float* b3  = (const float*)d_in[6];
  const float* pcw = (const float*)d_in[7];
  const float* pcb = (const float*)d_in[8];
  const float* ecw = (const float*)d_in[9];
  float* out = (float*)d_out;

  char* w = (char*)d_ws;
  size_t off = 0;
  _Float16* a2   = (_Float16*)(w + off); off += (size_t)4 * KB2_ * 32 * 16 * 2;  // 288 KB
  _Float16* a3   = (_Float16*)(w + off); off += (size_t)4 * KB3_ * 32 * 16 * 2;  // 16 KB
  _Float16* h16p = (_Float16*)(w + off); off += (size_t)B_ * F_ * CPP_ * 2;      // 10.08 MB (padded)
  _Float16* yr16 = (_Float16*)(w + off); off += (size_t)B_ * F_ * P_ * 2;        // 7.28 MB
  float*    h3   = (float*)(w + off);    off += (size_t)B_ * F_ * P_ * 4;        // 14.6 MB
  float*    u    = (float*)(w + off);    off += (size_t)B_ * N_ * 8 * 4;         // 14.6 MB
  _Float16* uh   = (_Float16*)(w + off); off += (size_t)B_ * N_ * E_ * OD_ * 2;  // 116.5 MB (L2-resident)
  float*    bl   = (float*)(w + off);    off += (size_t)B_ * N_ * E_ * 4;        // 14.6 MB
  float*    s    = (float*)(w + off);    off += 2048 * 4;
  float*    v    = (float*)(w + off);    off += 2048 * 4;
  (void)ws_size; (void)in_sizes; (void)n_in; (void)out_size;

  const int padDwords = (B_ * F_ * CPP_) / 2;    // 2,519,552 = 9842*256 exact
  k_prep<<<576, 256, 0, stream>>>(w2, a2, KB2_, K2_);
  k_prep<<<32, 256, 0, stream>>>(w3, a3, KB3_, K3_);
  k_zero<<<padDwords / 256, 256, 0, stream>>>((float*)h16p, padDwords);
  k_zero<<<(B_ * N_ * E_) / 256, 256, 0, stream>>>(bl, B_ * N_ * E_);
  k_conv1<<<(B_ * F_ * P_) / 256, 256, 0, stream>>>(x, w1, b1, h16p);
  k_conv2<<<1778, 256, 0, stream>>>(a2, h16p, b2, yr16);  // 14224 tiles / 8 waves
  k_conv3<<<1778, 256, 0, stream>>>(a3, h16p, yr16, b3, h3);
  k_pc<<<(B_ * N_) / 256, 256, 0, stream>>>(h3, pcw, pcb, u);
  k_uhat<<<N_ / 8, 256, 0, stream>>>(ecw, u, uh);
  for (int r = 0; r < 3; ++r) {
    k_zero<<<8, 256, 0, stream>>>(s, 2048);
    k_spart<<<B_ * 28, 256, 0, stream>>>(bl, uh, s);
    k_squash<<<1, 128, 0, stream>>>(s, v, out, r == 2 ? 1 : 0);
    if (r < 2) k_bupd<<<(B_ * N_ * E_) / 256, 256, 0, stream>>>(uh, v, bl);
  }
}
